// MLA_67061619359960
// MI455X (gfx1250) — matrix operand order repaired and checked by reading, not compiled, not run on hardware
//
#include <hip/hip_runtime.h>
#include <hip/hip_bf16.h>

typedef __bf16 bf16;
typedef __attribute__((ext_vector_type(16))) __bf16 bf16x16;
typedef __attribute__((ext_vector_type(8)))  __bf16 bf16x8;
typedef __attribute__((ext_vector_type(8)))  float  f32x8;
typedef __attribute__((ext_vector_type(4)))  int    i32x4;

#define NEG_INF_F (-3.0e38f)

__device__ __forceinline__ f32x8 wmma_bf16(bf16x16 a, bf16x16 b, f32x8 c) {
  return __builtin_amdgcn_wmma_f32_16x16x32_bf16(false, a, false, b, (short)0, c, false, false);
}

// A fragment: 16x32 bf16, row-major source with stride lda.
// Lane l holds row (l&15); K layout per ISA: half0 -> K {0..7,16..23}, half1 -> K {8..15,24..31}.
__device__ __forceinline__ bf16x16 load_a_frag(const bf16* base, int lda,
                                               int m0, int k0, int lane) {
  int row = lane & 15;
  int ks  = (lane & 16) ? 8 : 0;
  const bf16* p = base + (size_t)(m0 + row) * lda + k0 + ks;
  union { bf16x16 v; bf16x8 h[2]; } u;
  u.h[0] = *(const bf16x8*)(p);
  u.h[1] = *(const bf16x8*)(p + 16);
  return u.v;
}

// B fragment: 32x16 bf16 from an N-major (transposed) source Bt[n*ldb + k].
// Lane l holds col (l&15); half0 -> K {0..7,16..23}, half1 -> K {8..15,24..31}.
__device__ __forceinline__ bf16x16 load_b_frag(const bf16* base, int ldb,
                                               int n0, int k0, int lane) {
  int col = lane & 15;
  int ks  = (lane & 16) ? 8 : 0;
  const bf16* p = base + (size_t)(n0 + col) * ldb + k0 + ks;
  union { bf16x16 v; bf16x8 h[2]; } u;
  u.h[0] = *(const bf16x8*)(p);
  u.h[1] = *(const bf16x8*)(p + 16);
  return u.v;
}

// -------------------- async global -> LDS staging (gfx1250 ASYNCcnt path) ---
#if __has_builtin(__builtin_amdgcn_global_load_async_to_lds_b128) && \
    __has_builtin(__builtin_amdgcn_s_wait_asynccnt)
#define HAVE_ASYNC_LDS 1
#endif

__device__ __forceinline__ void copy16_g2lds(bf16* lds, const bf16* g) {
#ifdef HAVE_ASYNC_LDS
  __builtin_amdgcn_global_load_async_to_lds_b128(
      (__attribute__((address_space(1))) i32x4*)(g),
      (__attribute__((address_space(3))) i32x4*)(lds), 0, 0);
#else
  *(uint4*)lds = *(const uint4*)g;
#endif
}
__device__ __forceinline__ void async_g2lds_wait() {
#ifdef HAVE_ASYNC_LDS
  __builtin_amdgcn_s_wait_asynccnt(0);
#endif
}

// ---------------------------------------------------------------- utilities
__global__ void f32_to_bf16_kernel(const float* __restrict__ in, bf16* __restrict__ out, size_t n) {
  size_t i = (size_t)blockIdx.x * 256 + threadIdx.x;
  if (i < n) out[i] = (bf16)in[i];
}

// in: K x N fp32 row-major  ->  out: N x K bf16 row-major (LDS-tiled transpose)
__global__ void __launch_bounds__(256)
transpose_to_bf16(const float* __restrict__ in, bf16* __restrict__ out, int K, int N) {
  __shared__ float tile[32][33];
  int k0 = blockIdx.y * 32, n0 = blockIdx.x * 32;
  int tx = threadIdx.x & 31, ty = threadIdx.x >> 5;
  for (int i = 0; i < 32; i += 8) {
    int k = k0 + ty + i, n = n0 + tx;
    tile[ty + i][tx] = (k < K && n < N) ? in[(size_t)k * N + n] : 0.f;
  }
  __syncthreads();
  for (int i = 0; i < 32; i += 8) {
    int n = n0 + ty + i, k = k0 + tx;
    if (n < N && k < K) out[(size_t)n * K + k] = (bf16)tile[tx][ty + i];
  }
}

// rmsnorm over L elements starting at column `off` of a row with stride `stride_in`
__global__ void __launch_bounds__(256)
rmsnorm_bf16(const float* __restrict__ in, bf16* __restrict__ out,
             int stride_in, int off, int L) {
  __shared__ float red[8];
  int row = blockIdx.x;
  const float* x = in + (size_t)row * stride_in + off;
  float s = 0.f;
  for (int i = threadIdx.x; i < L; i += 256) { float v = x[i]; s += v * v; }
  for (int m = 16; m; m >>= 1) s += __shfl_xor(s, m, 32);
  if ((threadIdx.x & 31) == 0) red[threadIdx.x >> 5] = s;
  __syncthreads();
  s = red[0] + red[1] + red[2] + red[3] + red[4] + red[5] + red[6] + red[7];
  float r = rsqrtf(s / (float)L + 1e-6f);
  bf16* o = out + (size_t)row * L;
  for (int i = threadIdx.x; i < L; i += 256) o[i] = (bf16)(x[i] * r);
}

__device__ __forceinline__ float rope_val(const float* rowp, int d, int pos) {
  int j = d & 31;
  float x = rowp[d];
  float partner = rowp[(d < 32) ? d + 32 : d - 32];
  float rh = (d < 32) ? -partner : partner;
  float freq = __powf(10000.f, -(float)(2 * j) / 64.f);
  float ang = (float)pos * freq;
  return x * __cosf(ang) + rh * __sinf(ang);
}

// roped k_rope from ckv[:, 0:64] -> kr (rows x 64) bf16
__global__ void krope_kernel(const float* __restrict__ ckv, bf16* __restrict__ kr, int S) {
  int idx = blockIdx.x * 256 + threadIdx.x;      // row*64 + d
  int row = idx >> 6, d = idx & 63;
  int pos = row % S;
  kr[idx] = (bf16)rope_val(ckv + (size_t)row * 576, d, pos);
}

// qfull fp32 (rows x 3072) -> Qarr bf16 (B,H,S,192) with RoPE on last 64
__global__ void arrange_q_kernel(const float* __restrict__ qfull, bf16* __restrict__ Qarr, int S) {
  int idx = blockIdx.x * 256 + threadIdx.x;      // rows*3072
  int row = idx / 3072, c = idx % 3072;
  int h = c / 192, d = c % 192;
  int b = row / S, s = row % S;
  float outv;
  if (d < 128) outv = qfull[idx];
  else outv = rope_val(qfull + (size_t)row * 3072 + h * 192 + 128, d - 128, s);
  Qarr[((size_t)(b * 16 + h) * S + s) * 192 + d] = (bf16)outv;
}

// kvout fp32 (rows x 4096=H*256) + kr -> Karr (B,H,S,192), Vt (B,H,128,S)
__global__ void arrange_kv_kernel(const float* __restrict__ kvout, const bf16* __restrict__ kr,
                                  bf16* __restrict__ Karr, bf16* __restrict__ Vt, int S) {
  int idx = blockIdx.x * 256 + threadIdx.x;      // rows*4096
  int row = idx >> 12, c = idx & 4095;
  int h = c >> 8, d = c & 255;
  int b = row / S, s = row % S;
  int bh = b * 16 + h;
  float v = kvout[idx];
  if (d < 128) {
    Karr[((size_t)bh * S + s) * 192 + d] = (bf16)v;
    if (d < 64) Karr[((size_t)bh * S + s) * 192 + 128 + d] = kr[(size_t)row * 64 + d];
  } else {
    Vt[((size_t)bh * 128 + (d - 128)) * S + s] = (bf16)v;
  }
}

// ---------------------------------------------------------------- GEMM
// C[M,N] (fp32) = A[M,K](bf16, row-major) x Bt[N,K](bf16, N-major, padded to 128 cols) + bias[N]
// Block = 8 waves; each wave computes a 32x64 tile (block tile 128M x 128N).
__global__ void __launch_bounds__(256)
gemm_bf16_wmma(const bf16* __restrict__ A, const bf16* __restrict__ Bt,
               const float* __restrict__ bias, float* __restrict__ C,
               int M, int N, int K) {
  int lane = threadIdx.x & 31;
  int wave = threadIdx.x >> 5;
  int m0 = blockIdx.y * 128 + (wave & 3) * 32;
  int n0 = blockIdx.x * 128 + (wave >> 2) * 64;
  f32x8 acc[2][4] = {};
  for (int k0 = 0; k0 < K; k0 += 32) {
    bf16x16 a0 = load_a_frag(A, K, m0,      k0, lane);
    bf16x16 a1 = load_a_frag(A, K, m0 + 16, k0, lane);
    #pragma unroll
    for (int ni = 0; ni < 4; ++ni) {
      bf16x16 b = load_b_frag(Bt, K, n0 + ni * 16, k0, lane);
      acc[0][ni] = wmma_bf16(a0, b, acc[0][ni]);
      acc[1][ni] = wmma_bf16(a1, b, acc[1][ni]);
    }
  }
  int half = lane >> 4, cn = lane & 15;
  #pragma unroll
  for (int mi = 0; mi < 2; ++mi)
    #pragma unroll
    for (int ni = 0; ni < 4; ++ni) {
      int n = n0 + ni * 16 + cn;
      if (n < N) {
        float bv = bias[n];
        #pragma unroll
        for (int vg = 0; vg < 8; ++vg) {
          int m = m0 + mi * 16 + vg + half * 8;
          C[(size_t)m * N + n] = acc[mi][ni][vg] + bv;
        }
      }
    }
}

// ---------------------------------------------------------------- Attention
// Flash-style causal attention: one (b,h) + 128 q rows per block, 16 q rows / wave.
// K (64x192) and V (128x64) tiles staged in LDS via async global->LDS copies,
// shared by all 8 waves; P staged per-wave in LDS for the PxV A-fragment.
__global__ void __launch_bounds__(256)
mla_attention(const bf16* __restrict__ Qarr, const bf16* __restrict__ Karr,
              const bf16* __restrict__ Vt, bf16* __restrict__ attnbuf, int S) {
  __shared__ bf16 Kst[64][192];                   // 24 KB
  __shared__ bf16 Vst[128][64];                   // 16 KB
  __shared__ bf16 Pst[8][16][64];                 // 16 KB
  int lane = threadIdx.x & 31, wave = threadIdx.x >> 5;
  int bh = blockIdx.y;
  int b = bh >> 4, h = bh & 15;
  int q0 = blockIdx.x * 128 + wave * 16;
  const bf16* Q  = Qarr + (size_t)bh * S * 192;
  const bf16* Kp = Karr + (size_t)bh * S * 192;
  const bf16* V  = Vt   + (size_t)bh * 128 * S;

  bf16x16 qf[6];
  #pragma unroll
  for (int dk = 0; dk < 6; ++dk) qf[dk] = load_a_frag(Q, 192, q0, dk * 32, lane);

  f32x8 o[8] = {};
  float mrow[8], lrow[8];
  #pragma unroll
  for (int r = 0; r < 8; ++r) { mrow[r] = NEG_INF_F; lrow[r] = 0.f; }

  const float scale = 0.07216878364870323f;       // 1/sqrt(192)
  int half = lane >> 4, cn = lane & 15;
  int kb_end = (blockIdx.x * 128 + 191) / 64;     // block-wide causal bound
  for (int kb = 0; kb < kb_end; ++kb) {
    int kbase = kb * 64;
    __syncthreads();                              // previous tile fully consumed
    // --- cooperative stage: K tile 64x192 (1536 x 16B), V tile 128x64 (1024 x 16B)
    #pragma unroll
    for (int i = 0; i < 6; ++i) {
      int chunk = threadIdx.x + i * 256;
      int r = chunk / 24, c = (chunk % 24) * 8;
      copy16_g2lds(&Kst[r][c], Kp + (size_t)(kbase + r) * 192 + c);
    }
    #pragma unroll
    for (int i = 0; i < 4; ++i) {
      int chunk = threadIdx.x + i * 256;
      int r = chunk >> 3, c = (chunk & 7) * 8;
      copy16_g2lds(&Vst[r][c], V + (size_t)r * S + kbase + c);
    }
    async_g2lds_wait();
    __syncthreads();

    if (kbase <= q0 + 15) {                       // this wave's causal range
      f32x8 sacc[4] = {};
      #pragma unroll
      for (int dk = 0; dk < 6; ++dk)
        #pragma unroll
        for (int j = 0; j < 4; ++j) {
          bf16x16 kf = load_b_frag(&Kst[0][0], 192, j * 16, dk * 32, lane);
          sacc[j] = wmma_bf16(qf[dk], kf, sacc[j]);
        }
      // scale + causal mask
      #pragma unroll
      for (int j = 0; j < 4; ++j) {
        int col = kbase + j * 16 + cn;
        #pragma unroll
        for (int vg = 0; vg < 8; ++vg) {
          int qpos = q0 + vg + half * 8;
          float sv = sacc[j][vg] * scale;
          sacc[j][vg] = (col <= qpos) ? sv : NEG_INF_F;
        }
      }
      // online softmax (rows live in 16-lane halves of the C layout)
      #pragma unroll
      for (int vg = 0; vg < 8; ++vg) {
        float v = fmaxf(fmaxf(sacc[0][vg], sacc[1][vg]), fmaxf(sacc[2][vg], sacc[3][vg]));
        v = fmaxf(v, __shfl_xor(v, 1)); v = fmaxf(v, __shfl_xor(v, 2));
        v = fmaxf(v, __shfl_xor(v, 4)); v = fmaxf(v, __shfl_xor(v, 8));
        float mnew = fmaxf(mrow[vg], v);
        float cf = __expf(mrow[vg] - mnew);
        mrow[vg] = mnew;
        lrow[vg] *= cf;
        #pragma unroll
        for (int nt = 0; nt < 8; ++nt) o[nt][vg] *= cf;
        float rs = 0.f;
        #pragma unroll
        for (int j = 0; j < 4; ++j) {
          float p = __expf(sacc[j][vg] - mnew);
          sacc[j][vg] = p;
          rs += p;
        }
        rs += __shfl_xor(rs, 1); rs += __shfl_xor(rs, 2);
        rs += __shfl_xor(rs, 4); rs += __shfl_xor(rs, 8);
        lrow[vg] += rs;
      }
      // stage P (C layout -> row-major bf16 in per-wave LDS)
      #pragma unroll
      for (int j = 0; j < 4; ++j)
        #pragma unroll
        for (int vg = 0; vg < 8; ++vg)
          Pst[wave][vg + half * 8][j * 16 + cn] = (bf16)sacc[j][vg];
      // P (16x64) x V (64x128), split K into two 32-chunks
      #pragma unroll
      for (int kk = 0; kk < 2; ++kk) {
        int prow = lane & 15;
        int ks = (lane & 16) ? 8 : 0;
        union { bf16x16 v; bf16x8 hh[2]; } u;
        u.hh[0] = *(const bf16x8*)&Pst[wave][prow][kk * 32 + ks];
        u.hh[1] = *(const bf16x8*)&Pst[wave][prow][kk * 32 + 16 + ks];
        bf16x16 pf = u.v;
        #pragma unroll
        for (int nt = 0; nt < 8; ++nt) {
          bf16x16 vf = load_b_frag(&Vst[0][0], 64, nt * 16, kk * 32, lane);
          o[nt] = wmma_bf16(pf, vf, o[nt]);
        }
      }
    }
  }
  // epilogue: divide by row sum, write (b, s, h*128 + dv) bf16
  #pragma unroll
  for (int vg = 0; vg < 8; ++vg) {
    float inv = 1.f / lrow[vg];
    int qpos = q0 + vg + half * 8;
    bf16* outp = attnbuf + ((size_t)b * S + qpos) * 2048 + h * 128;
    #pragma unroll
    for (int nt = 0; nt < 8; ++nt)
      outp[nt * 16 + cn] = (bf16)(o[nt][vg] * inv);
  }
}

// ---------------------------------------------------------------- launcher
extern "C" void kernel_launch(void* const* d_in, const int* in_sizes, int n_in,
                              void* d_out, int out_size, void* d_ws, size_t ws_size,
                              hipStream_t stream) {
  (void)in_sizes; (void)n_in; (void)out_size; (void)ws_size;
  const float* X        = (const float*)d_in[0];
  const float* Wq_down  = (const float*)d_in[2];
  const float* bq_down  = (const float*)d_in[3];
  const float* Wq_up    = (const float*)d_in[4];
  const float* bq_up    = (const float*)d_in[5];
  const float* Wkv_down = (const float*)d_in[6];
  const float* bkv_down = (const float*)d_in[7];
  const float* Wkv_up   = (const float*)d_in[8];
  const float* bkv_up   = (const float*)d_in[9];
  const float* Wo       = (const float*)d_in[10];
  const float* bo       = (const float*)d_in[11];
  float* out = (float*)d_out;

  const int S = 2048, HID = 2048, QLR = 1536, KVLR = 512;
  const int ROWS = 2 * S;                         // 4096
  const int NQ = 16 * 192;                        // 3072
  const int NKV = 16 * 256;                       // 4096
  const int NKVD = 576, NKVD_PAD = 640;           // kv_down N padded to 128 multiple

  char* ws = (char*)d_ws;
  size_t off = 0;
  auto alloc = [&](size_t bytes) {
    off = (off + 255) & ~(size_t)255;
    char* p = ws + off;
    off += bytes;
    return p;
  };
  bf16*  Xbf    = (bf16*)alloc((size_t)ROWS * HID * 2);
  bf16*  Wqd_t  = (bf16*)alloc((size_t)QLR * HID * 2);
  bf16*  Wqu_t  = (bf16*)alloc((size_t)NQ * QLR * 2);
  bf16*  Wkvd_t = (bf16*)alloc((size_t)NKVD_PAD * HID * 2);
  bf16*  Wkvu_t = (bf16*)alloc((size_t)NKV * KVLR * 2);
  bf16*  Wo_t   = (bf16*)alloc((size_t)HID * HID * 2);
  bf16*  qdownn = (bf16*)alloc((size_t)ROWS * QLR * 2);
  bf16*  ckvn   = (bf16*)alloc((size_t)ROWS * KVLR * 2);
  bf16*  kr     = (bf16*)alloc((size_t)ROWS * 64 * 2);
  bf16*  Qarr   = (bf16*)alloc((size_t)32 * S * 192 * 2);
  bf16*  Karr   = (bf16*)alloc((size_t)32 * S * 192 * 2);
  bf16*  Vt     = (bf16*)alloc((size_t)32 * 128 * S * 2);
  bf16*  attnb  = (bf16*)alloc((size_t)ROWS * HID * 2);
  float* scr    = (float*)alloc((size_t)ROWS * NKV * 4);   // reused: qdown/qfull/ckv/kvout

  dim3 blk(256);
  // 1. convert X to bf16
  f32_to_bf16_kernel<<<(ROWS * HID) / 256, blk, 0, stream>>>(X, Xbf, (size_t)ROWS * HID);
  // 2. transpose weights to N x K bf16
  transpose_to_bf16<<<dim3(QLR / 32, HID / 32), blk, 0, stream>>>(Wq_down, Wqd_t, HID, QLR);
  transpose_to_bf16<<<dim3(NQ / 32, QLR / 32), blk, 0, stream>>>(Wq_up, Wqu_t, QLR, NQ);
  transpose_to_bf16<<<dim3(NKVD / 32, HID / 32), blk, 0, stream>>>(Wkv_down, Wkvd_t, HID, NKVD);
  transpose_to_bf16<<<dim3(NKV / 32, KVLR / 32), blk, 0, stream>>>(Wkv_up, Wkvu_t, KVLR, NKV);
  transpose_to_bf16<<<dim3(HID / 32, HID / 32), blk, 0, stream>>>(Wo, Wo_t, HID, HID);
  // 3. q_down = X @ Wq_down + b     (scr = ROWS x QLR fp32)
  gemm_bf16_wmma<<<dim3(QLR / 128, ROWS / 128), blk, 0, stream>>>(Xbf, Wqd_t, bq_down, scr, ROWS, QLR, HID);
  // 4. rmsnorm(q_down) -> qdownn bf16
  rmsnorm_bf16<<<ROWS, blk, 0, stream>>>(scr, qdownn, QLR, 0, QLR);
  // 5. q = qdownn @ Wq_up + b       (scr = ROWS x 3072 fp32)
  gemm_bf16_wmma<<<dim3(NQ / 128, ROWS / 128), blk, 0, stream>>>(qdownn, Wqu_t, bq_up, scr, ROWS, NQ, QLR);
  // 6. split heads + RoPE -> Qarr
  arrange_q_kernel<<<(ROWS * NQ) / 256, blk, 0, stream>>>(scr, Qarr, S);
  // 7. ckv = X @ Wkv_down + b       (scr = ROWS x 576 fp32)
  gemm_bf16_wmma<<<dim3(NKVD_PAD / 128, ROWS / 128), blk, 0, stream>>>(Xbf, Wkvd_t, bkv_down, scr, ROWS, NKVD, HID);
  // 8. rmsnorm(ckv[:,64:]) -> ckvn ; rope(ckv[:,:64]) -> kr
  rmsnorm_bf16<<<ROWS, blk, 0, stream>>>(scr, ckvn, 576, 64, KVLR);
  krope_kernel<<<(ROWS * 64) / 256, blk, 0, stream>>>(scr, kr, S);
  // 9. kv = ckvn @ Wkv_up + b       (scr = ROWS x 4096 fp32)
  gemm_bf16_wmma<<<dim3(NKV / 128, ROWS / 128), blk, 0, stream>>>(ckvn, Wkvu_t, bkv_up, scr, ROWS, NKV, KVLR);
  // 10. split k_nope/value, merge rope -> Karr, Vt
  arrange_kv_kernel<<<(ROWS * NKV) / 256, blk, 0, stream>>>(scr, kr, Karr, Vt, S);
  // 11. flash attention -> attnb bf16 (rows x H*DV)
  mla_attention<<<dim3(S / 128, 32), blk, 0, stream>>>(Qarr, Karr, Vt, attnb, S);
  // 12. out = attn @ Wo + bo (fp32)
  gemm_bf16_wmma<<<dim3(HID / 128, ROWS / 128), blk, 0, stream>>>(attnb, Wo_t, bo, out, ROWS, HID, HID);
}
